// GATModel_47820165874392
// MI455X (gfx1250) — matrix operand order repaired and checked by reading, not compiled, not run on hardware
//
#include <hip/hip_runtime.h>
#include <hip/hip_bf16.h>

// ---------------- problem constants (from reference) ----------------
#define Nn        50000
#define Ee        400000
#define F_IN      128
#define Hh        8
#define Cc        32
#define HCv       256
#define EDGE_DIM  26
#define Gg        50
#define NUM_CLS   151
#define NEG_SLOPE 0.2f

typedef __attribute__((ext_vector_type(16))) _Float16 v16h;
typedef __attribute__((ext_vector_type(8)))  float    v8f;

// ---- order-preserving float<->uint mapping for atomicMax-based segment max ----
__device__ __forceinline__ unsigned f2ord(float f) {
    unsigned u = __float_as_uint(f);
    return (u & 0x80000000u) ? ~u : (u | 0x80000000u);
}
__device__ __forceinline__ float ord2f(unsigned v) {
    unsigned u = (v & 0x80000000u) ? (v & 0x7FFFFFFFu) : ~v;
    return __uint_as_float(u);
}
#define ORD_NEG_INF 0x007FFFFFu   // f2ord(-inf)

// =====================================================================
// GEMM with bias: out[M,256] = A[M,K] @ W[K,256] + b   (f16 WMMA, f32 acc)
//
// Block = 256 threads (8 waves) -> 128-row x 64-col output macro-tile.
//  * W[:, n0:n0+64] staged once per block into LDS as f16, pre-swizzled
//    into WMMA B-fragment order: each (k-chunk, tile, lane) fragment is a
//    contiguous, 32B-aligned v16h -> one vector LDS read (2x ds_load_b128)
//    instead of 16x ds_load_u16.
//  * Each wave: 16x64 strip = 4 WMMA accumulators; one A fragment per
//    K-step reused across 4 v_wmma issues.
//  * __builtin_prefetch on next A chunk -> global_prefetch_b8.
// K % 32 == 0 and M % 16 == 0 guaranteed (K in {128,256}, M=50000).
// =====================================================================
#define GB_COLS 64                      // columns per block (4 WMMA tiles)
__global__ void gemm_bias_wmma(const float* __restrict__ A,
                               const float* __restrict__ W,
                               const float* __restrict__ bias,
                               float* __restrict__ out,
                               int M, int K) {
    // fragment-major B panel: [K/32][4 tiles][32 lanes] x 16 halfs = K*64 halfs
    __shared__ v16h sWf[(HCv >> 5) * 4 * 32];       // max K=256 -> 1024 frags = 32 KB

    const int lane = threadIdx.x & 31;
    const int wave = threadIdx.x >> 5;
    const int colBlk = blockIdx.x & 3;              // 256/64 = 4 col blocks
    const int rowBlk = blockIdx.x >> 2;
    const int n0 = colBlk * GB_COLS;

    // ---- stage W panel into LDS, swizzled to B-fragment layout ----
    // fragment f = (kc*4 + t)*32 + l ; element i of lane l's fragment is
    // W[kc*32 + (l<16?0:8) + (i<8 ? i : i+8)][n0 + t*16 + (l&15)]  (ISA 7.12.2 B layout)
    {
        _Float16* sW = (_Float16*)sWf;
        const int nfrag = (K >> 5) * 4 * 32;
        for (int f = threadIdx.x; f < nfrag; f += blockDim.x) {
            const int l  = f & 31;
            const int t  = (f >> 5) & 3;
            const int kc = f >> 7;
            const int kb = (kc << 5) + ((l < 16) ? 0 : 8);
            const int col = n0 + t * 16 + (l & 15);
            _Float16* dp = sW + ((size_t)f << 4);
            const float* sp = W + (size_t)kb * HCv + col;
            #pragma unroll
            for (int i = 0; i < 16; ++i)
                dp[i] = (_Float16)sp[(size_t)(i < 8 ? i : i + 8) * HCv];
        }
    }
    __syncthreads();

    const int r0 = rowBlk * 128 + wave * 16;        // wave's 16-row strip
    if (r0 >= M) return;                            // wave-uniform: EXEC stays full

    // A fragment addressing (ISA 7.12.2): lanes 0-15 K 0-7/16-23, lanes 16-31 K 8-15/24-31
    const int arow = lane & 15;
    const int akb  = (lane < 16) ? 0 : 8;
    const int moff = (lane < 16) ? 0 : 8;           // C/D: VGPR j <-> row j + moff
    const int ncol = lane & 15;

    v8f c0 = {}, c1 = {}, c2 = {}, c3 = {};
    const float* arp = A + (size_t)(r0 + arow) * K;

    const int kchunks = K >> 5;
    for (int kc = 0; kc < kchunks; ++kc) {
        // A fragment (f32 -> f16), reused by all 4 column tiles
        v16h a;
        const float* ap = arp + (kc << 5) + akb;
        #pragma unroll
        for (int i = 0; i < 8; ++i) a[i]     = (_Float16)ap[i];
        #pragma unroll
        for (int i = 0; i < 8; ++i) a[8 + i] = (_Float16)ap[16 + i];
        if (kc + 1 < kchunks) __builtin_prefetch(ap + 32, 0, 0);    // global_prefetch_b8

        // B fragments: single aligned v16h LDS reads (fragment-major layout)
        const int fb = (kc << 2) * 32 + lane;
        v16h b0 = sWf[fb];
        v16h b1 = sWf[fb + 32];
        v16h b2 = sWf[fb + 64];
        v16h b3 = sWf[fb + 96];
        c0 = __builtin_amdgcn_wmma_f32_16x16x32_f16(false, a, false, b0, (short)0, c0, false, false);
        c1 = __builtin_amdgcn_wmma_f32_16x16x32_f16(false, a, false, b1, (short)0, c1, false, false);
        c2 = __builtin_amdgcn_wmma_f32_16x16x32_f16(false, a, false, b2, (short)0, c2, false, false);
        c3 = __builtin_amdgcn_wmma_f32_16x16x32_f16(false, a, false, b3, (short)0, c3, false, false);
    }

    // ---- store 4 tiles + bias ----
    float* orow = out + (size_t)(r0 + moff) * HCv + n0 + ncol;
    const float bv0 = bias[n0 + ncol];
    const float bv1 = bias[n0 + 16 + ncol];
    const float bv2 = bias[n0 + 32 + ncol];
    const float bv3 = bias[n0 + 48 + ncol];
    #pragma unroll
    for (int j = 0; j < 8; ++j) {
        float* op = orow + (size_t)j * HCv;
        op[0]  = c0[j] + bv0;
        op[16] = c1[j] + bv1;
        op[32] = c2[j] + bv2;
        op[48] = c3[j] + bv3;
    }
}

// =====================================================================
// Edge attention logits: fused edge-MLP (We staged in LDS) + leaky_relu +
// dot with att + wave reduction; atomicMax segment-max per (dst, head).
// One wave per edge (grid-stride).
// =====================================================================
__global__ void edge_logits_kernel(const float* __restrict__ xl,
                                   const float* __restrict__ xr,
                                   const int* __restrict__ src,
                                   const int* __restrict__ dst,
                                   const float* __restrict__ eattr,
                                   const float* __restrict__ We,   // [26,256]
                                   const float* __restrict__ att,  // [256]
                                   float* __restrict__ logits,     // [E,8]
                                   unsigned* __restrict__ nodemax) // [N,8]
{
    __shared__ float sWe[EDGE_DIM * HCv];
    __shared__ float sAtt[HCv];
    for (int i = threadIdx.x; i < EDGE_DIM * HCv; i += blockDim.x) sWe[i] = We[i];
    for (int i = threadIdx.x; i < HCv; i += blockDim.x) sAtt[i] = att[i];
    __syncthreads();

    const int lane  = threadIdx.x & 31;
    const int wave  = threadIdx.x >> 5;
    const int wpg   = gridDim.x * (blockDim.x >> 5);
    for (int e = blockIdx.x * (blockDim.x >> 5) + wave; e < Ee; e += wpg) {
        const int s = src[e], d = dst[e];
        const float ea = (lane < EDGE_DIM) ? eattr[(size_t)e * EDGE_DIM + lane] : 0.f;
        const float* xls = xl + (size_t)s * HCv;
        const float* xrd = xr + (size_t)d * HCv;
        #pragma unroll
        for (int h = 0; h < Hh; ++h) {
            const int hc = h * Cc + lane;
            float ev = 0.f;
            #pragma unroll
            for (int k = 0; k < EDGE_DIM; ++k)
                ev += __shfl(ea, k, 32) * sWe[k * HCv + hc];
            float z = xls[hc] + xrd[hc] + ev;
            z = (z > 0.f) ? z : NEG_SLOPE * z;          // GATv2: lrelu before dot
            float p = z * sAtt[hc];
            #pragma unroll
            for (int off = 16; off; off >>= 1) p += __shfl_xor(p, off, 32);
            if (lane == 0) {
                logits[(size_t)e * Hh + h] = p;
                atomicMax(nodemax + (size_t)d * Hh + h, f2ord(p));
            }
        }
    }
}

// =====================================================================
// ex = exp(logit - max[dst]); in-place; atomicAdd into denom[dst,h]
// =====================================================================
__global__ void edge_exp_kernel(float* __restrict__ lg,            // [E,8] in/out
                                const unsigned* __restrict__ nodemax,
                                const int* __restrict__ dst,
                                float* __restrict__ denom)         // [N,8]
{
    const int t = blockIdx.x * blockDim.x + threadIdx.x;
    if (t >= Ee * Hh) return;
    const int e = t >> 3, h = t & 7;
    const int d = dst[e];
    const float m  = ord2f(nodemax[(size_t)d * Hh + h]);
    const float ex = __expf(lg[t] - m);
    lg[t] = ex;
    atomicAdd(denom + (size_t)d * Hh + h, ex);
}

// =====================================================================
// msg scatter: acc[dst] += xl[src] * alpha ; one wave per edge
// =====================================================================
__global__ void edge_scatter_kernel(const float* __restrict__ xl,
                                    const float* __restrict__ ex,    // [E,8]
                                    const float* __restrict__ denom, // [N,8]
                                    const int* __restrict__ src,
                                    const int* __restrict__ dst,
                                    float* __restrict__ acc)         // [N,256]
{
    const int lane = threadIdx.x & 31;
    const int wave = threadIdx.x >> 5;
    const int wpg  = gridDim.x * (blockDim.x >> 5);
    for (int e = blockIdx.x * (blockDim.x >> 5) + wave; e < Ee; e += wpg) {
        const int s = src[e], d = dst[e];
        float a = 0.f;
        if (lane < Hh)
            a = ex[(size_t)e * Hh + lane] / (denom[(size_t)d * Hh + lane] + 1e-16f);
        const float* xls = xl + (size_t)s * HCv;
        float* accd = acc + (size_t)d * HCv;
        #pragma unroll
        for (int h = 0; h < Hh; ++h) {
            const float alpha = __shfl(a, h, 32);
            const int hc = h * Cc + lane;
            atomicAdd(accd + hc, xls[hc] * alpha);
        }
    }
}

// out = relu(acc + bias)
__global__ void bias_relu_kernel(const float* __restrict__ acc,
                                 const float* __restrict__ bias,
                                 float* __restrict__ out, int n) {
    const int t = blockIdx.x * blockDim.x + threadIdx.x;
    if (t < n) {
        float v = acc[t] + bias[t & (HCv - 1)];
        out[t] = (v > 0.f) ? v : 0.f;
    }
}

__global__ void fill_u32_kernel(unsigned* __restrict__ p, unsigned v, int n) {
    const int t = blockIdx.x * blockDim.x + threadIdx.x;
    if (t < n) p[t] = v;
}

// global mean-pool accumulation
__global__ void pool_kernel(const float* __restrict__ h,
                            const int* __restrict__ batch,
                            float* __restrict__ pooled,   // [G,256]
                            float* __restrict__ cnt) {    // [G]
    const int t = blockIdx.x * blockDim.x + threadIdx.x;
    if (t >= Nn * HCv) return;
    const int n = t >> 8, hc = t & (HCv - 1);
    const int g = batch[n];
    atomicAdd(pooled + (size_t)g * HCv + hc, h[t]);
    if (hc == 0) atomicAdd(cnt + g, 1.0f);
}

// out[g,j] = (pooled[g]/max(cnt,1)) @ Wfc + bfc
__global__ void final_fc_kernel(const float* __restrict__ pooled,
                                const float* __restrict__ cnt,
                                const float* __restrict__ Wfc,  // [256,151]
                                const float* __restrict__ bfc,
                                float* __restrict__ out) {
    const int t = blockIdx.x * blockDim.x + threadIdx.x;
    if (t >= Gg * NUM_CLS) return;
    const int g = t / NUM_CLS, j = t - g * NUM_CLS;
    const float inv = 1.0f / fmaxf(cnt[g], 1.0f);
    float s = 0.f;
    #pragma unroll 8
    for (int k = 0; k < HCv; ++k)
        s += pooled[(size_t)g * HCv + k] * Wfc[(size_t)k * NUM_CLS + j];
    out[t] = s * inv + bfc[j];
}

// =====================================================================
extern "C" void kernel_launch(void* const* d_in, const int* in_sizes, int n_in,
                              void* d_out, int out_size, void* d_ws, size_t ws_size,
                              hipStream_t stream) {
    const float* x     = (const float*)d_in[0];
    const int*   esrc  = (const int*)  d_in[1];
    const int*   edst  = (const int*)  d_in[2];
    const float* eattr = (const float*)d_in[3];
    const int*   batch = (const int*)  d_in[4];
    const float* Wl1 = (const float*)d_in[5],  *bl1 = (const float*)d_in[6];
    const float* Wr1 = (const float*)d_in[7],  *br1 = (const float*)d_in[8];
    const float* We1 = (const float*)d_in[9],  *att1 = (const float*)d_in[10];
    const float* bias1 = (const float*)d_in[11];
    const float* Wl2 = (const float*)d_in[12], *bl2 = (const float*)d_in[13];
    const float* Wr2 = (const float*)d_in[14], *br2 = (const float*)d_in[15];
    const float* We2 = (const float*)d_in[16], *att2 = (const float*)d_in[17];
    const float* bias2 = (const float*)d_in[18];
    const float* Wfc = (const float*)d_in[19], *bfc = (const float*)d_in[20];
    float* out = (float*)d_out;

    // ---------------- workspace carve-up (all fp32) ----------------
    float* xl   = (float*)d_ws;                 // N*256
    float* xr   = xl  + (size_t)Nn * HCv;       // N*256
    float* acc  = xr  + (size_t)Nn * HCv;       // N*256
    float* h1   = acc + (size_t)Nn * HCv;       // N*256 (layer output, reused)
    float* exb  = h1  + (size_t)Nn * HCv;       // E*8 logits -> exp (in place)
    unsigned* nodemax = (unsigned*)(exb + (size_t)Ee * Hh);  // N*8
    float* denom  = (float*)(nodemax + (size_t)Nn * Hh);     // N*8
    float* pooled = denom + (size_t)Nn * Hh;    // G*256
    float* cnt    = pooled + (size_t)Gg * HCv;  // G

    const int TPB = 256;
    const int rowBlks     = (Nn + 127) / 128;            // 128 rows per block
    const int gemm_blocks = rowBlks * (HCv / GB_COLS);   // x4 column blocks
    const int edge_blocks = 2048;                        // grid-stride, 8 waves/block
    const int eh_blocks   = (Ee * Hh + TPB - 1) / TPB;
    const int nhc_blocks  = (Nn * HCv + TPB - 1) / TPB;
    const int nh_blocks   = (Nn * Hh + TPB - 1) / TPB;

    for (int layer = 0; layer < 2; ++layer) {
        const float *Wl, *bl, *Wr, *br, *We, *att, *bias, *in;
        int K;
        if (layer == 0) { in = x;  K = F_IN; Wl=Wl1; bl=bl1; Wr=Wr1; br=br1; We=We1; att=att1; bias=bias1; }
        else            { in = h1; K = HCv;  Wl=Wl2; bl=bl2; Wr=Wr2; br=br2; We=We2; att=att2; bias=bias2; }

        // per-layer resets
        hipMemsetAsync(acc,   0, (size_t)Nn * HCv * sizeof(float), stream);
        hipMemsetAsync(denom, 0, (size_t)Nn * Hh  * sizeof(float), stream);
        fill_u32_kernel<<<nh_blocks, TPB, 0, stream>>>(nodemax, ORD_NEG_INF, Nn * Hh);

        // dense node transforms (WMMA)
        gemm_bias_wmma<<<gemm_blocks, TPB, 0, stream>>>(in, Wl, bl, xl, Nn, K);
        gemm_bias_wmma<<<gemm_blocks, TPB, 0, stream>>>(in, Wr, br, xr, Nn, K);

        // attention
        edge_logits_kernel<<<edge_blocks, TPB, 0, stream>>>(xl, xr, esrc, edst, eattr,
                                                            We, att, exb, nodemax);
        edge_exp_kernel<<<eh_blocks, TPB, 0, stream>>>(exb, nodemax, edst, denom);
        edge_scatter_kernel<<<edge_blocks, TPB, 0, stream>>>(xl, exb, denom, esrc, edst, acc);

        // out = relu(acc + bias) -> h1
        bias_relu_kernel<<<nhc_blocks, TPB, 0, stream>>>(acc, bias, h1, Nn * HCv);
    }

    // global mean pool + final linear
    hipMemsetAsync(pooled, 0, ((size_t)Gg * HCv + Gg) * sizeof(float), stream);
    pool_kernel<<<nhc_blocks, TPB, 0, stream>>>(h1, batch, pooled, cnt);
    final_fc_kernel<<<(Gg * NUM_CLS + TPB - 1) / TPB, TPB, 0, stream>>>(pooled, cnt, Wfc, bfc, out);
}
